// Graph_28973849379305
// MI455X (gfx1250) — matrix operand order repaired and checked by reading, not compiled, not run on hardware
//
#include <hip/hip_runtime.h>

// ---------------- problem constants (match reference) ----------------
#define NU_ 600000
#define NI_ 300000
#define NN_ (NU_ + NI_)
#define D_  64
#define DF_ 128
#define E_  3000000
#define EI_ 900000

typedef __attribute__((ext_vector_type(16))) __bf16 v16bf;
typedef __attribute__((ext_vector_type(8)))  __bf16 v8bf;
typedef __attribute__((ext_vector_type(4)))  __bf16 v4bf;
typedef __attribute__((ext_vector_type(8)))  float  v8f;

// ---------------- agg = scale * h0 (accumulator init, replaces zero+blend) --
template<int DIM>
__global__ __launch_bounds__(256) void init_kernel(
    float* __restrict__ agg, const float* __restrict__ h0u,
    const float* __restrict__ h0i, int split, int n, float scale)
{
    constexpr int CH = DIM / 4;
    size_t t = (size_t)blockIdx.x * blockDim.x + threadIdx.x;
    if (t >= (size_t)n * CH) return;
    int node = (int)(t / CH);
    int c = (int)(t % CH);
    const float* h0 = (node < split) ? (h0u + (size_t)node * DIM)
                                     : (h0i + (size_t)(node - split) * DIM);
    float4 h = *(const float4*)(h0 + c * 4);
    float4 o = make_float4(h.x * scale, h.y * scale, h.z * scale, h.w * scale);
    *(float4*)(agg + t * 4) = o;
}

// ---------------- 128-d scatter: one wave per edge (scalar edge metadata) ---
__global__ __launch_bounds__(256) void scatter128_kernel(
    const float* __restrict__ h, const int* __restrict__ src,
    const int* __restrict__ dst, const float* __restrict__ w,
    float* __restrict__ agg, int nE)
{
    int wv = blockIdx.x * 8 + (threadIdx.x >> 5);   // wave-uniform edge id
    if (wv >= nE) return;
    int lane = threadIdx.x & 31;
    int s = src[wv], d = dst[wv];                   // wave-uniform -> s_load
    float ww = w[wv];
    float4 v = *(const float4*)(h + (size_t)s * 128 + lane * 4);
    float* o = agg + (size_t)d * 128 + lane * 4;
    atomicAdd(o + 0, v.x * ww);
    atomicAdd(o + 1, v.y * ww);
    atomicAdd(o + 2, v.z * ww);
    atomicAdd(o + 3, v.w * ww);
}

// ---------------- 64-d scatter with split source (bipartite graph) ----------
__global__ __launch_bounds__(256) void scatter64_kernel(
    const float* __restrict__ hu, const float* __restrict__ hi, int split,
    const int* __restrict__ src, const int* __restrict__ dst,
    const float* __restrict__ w, float* __restrict__ agg, int nE)
{
    size_t t = (size_t)blockIdx.x * blockDim.x + threadIdx.x;
    if (t >= (size_t)nE * 16) return;
    int e = (int)(t >> 4);
    int c = (int)(t & 15);
    int s = src[e];
    const float* base = (s < split) ? (hu + (size_t)s * 64)
                                    : (hi + (size_t)(s - split) * 64);
    float4 v = *(const float4*)(base + c * 4);
    float ww = w[e];
    float* o = agg + (size_t)dst[e] * 64 + c * 4;
    atomicAdd(o + 0, v.x * ww);
    atomicAdd(o + 1, v.y * ww);
    atomicAdd(o + 2, v.z * ww);
    atomicAdd(o + 3, v.w * ww);
}

// ---------------- wp = s * w (edge-weight prescale) -------------------------
__global__ __launch_bounds__(256) void scalew_kernel(
    const float* __restrict__ w, float* __restrict__ wp, float s, int n)
{
    int t = blockIdx.x * blockDim.x + threadIdx.x;
    if (t < n) wp[t] = w[t] * s;
}

// ---------------- W[k][n] f32 -> Wt[n][k] bf16 (B-fragment friendly) --------
__global__ __launch_bounds__(256) void wtrans_kernel(
    const float* __restrict__ W, __bf16* __restrict__ Wt, int K, int N)
{
    int t = blockIdx.x * blockDim.x + threadIdx.x;
    if (t >= K * N) return;
    int k = t / N, n = t % N;
    Wt[(size_t)n * K + k] = (__bf16)W[t];
}

// ---------------- fused MLP: relu(relu(as_scale*[h0|h1] @ W1 + b1) @ W2 + b2)
// block = 512 threads = 16 waves; 16 rows/block.
// Layer1: wave w -> cols [16w,16w+16), K=256 = 8x wmma_f32_16x16x32_bf16.
// Layer2: waves 0..3 -> 64 cols, A from LDS hidden.
__global__ __launch_bounds__(512) void mlp_kernel(
    const float* __restrict__ h0, const float* __restrict__ h1, float as_scale,
    const __bf16* __restrict__ W1t, const float* __restrict__ b1,
    const __bf16* __restrict__ W2t, const float* __restrict__ b2,
    float* __restrict__ z)
{
    __shared__ __bf16 As[16][256];
    __shared__ __bf16 Hs[16][256];

    const int tid  = threadIdx.x;
    const int wave = tid >> 5;
    const int lane = tid & 31;
    const int row0 = blockIdx.x * 16;

    // vectorized bf16 staging of concat(h0,h1)*as_scale (ds_store_b64)
    for (int i = tid * 4; i < 16 * 256; i += 512 * 4) {
        int r = i >> 8, k = i & 255;
        const float* sp = (k < DF_) ? (h0 + (size_t)(row0 + r) * DF_ + k)
                                    : (h1 + (size_t)(row0 + r) * DF_ + (k - DF_));
        float4 v = *(const float4*)sp;
        v4bf o = { (__bf16)(v.x * as_scale), (__bf16)(v.y * as_scale),
                   (__bf16)(v.z * as_scale), (__bf16)(v.w * as_scale) };
        *(v4bf*)&As[r][k] = o;
    }
    __syncthreads();

    const int m  = lane & 15;
    const int hi = lane >> 4;
    const int kb = hi ? 16 : 0;

    // ---- layer 1 ----
    {
        const int ncol = wave * 16 + m;
        const __bf16* wrow = W1t + (size_t)ncol * 256;
        v8f acc = {};
        #pragma unroll 4
        for (int kk = 0; kk < 8; ++kk) {
            const int k0 = kk * 32;
            v8bf alo = *(const v8bf*)&As[m][k0 + hi * 8];        // ds_load_b128
            v8bf ahi = *(const v8bf*)&As[m][k0 + 16 + hi * 8];   // ds_load_b128
            v16bf a = __builtin_shufflevector(alo, ahi,
                0,1,2,3,4,5,6,7,8,9,10,11,12,13,14,15);
            v16bf b = __builtin_shufflevector(*(const v8bf*)(wrow + k0 + hi * 8),
                *(const v8bf*)(wrow + k0 + 16 + hi * 8),
                0,1,2,3,4,5,6,7,8,9,10,11,12,13,14,15);           // global_load_b128 x2
            acc = __builtin_amdgcn_wmma_f32_16x16x32_bf16(
                false, a, false, b, (short)0, acc, false, false);
        }
        const float bias = b1[ncol];
        #pragma unroll
        for (int j = 0; j < 8; ++j) {
            float v = acc[j] + bias;
            Hs[j + 8 * hi][ncol] = (__bf16)(v > 0.f ? v : 0.f);
        }
    }
    __syncthreads();

    // ---- layer 2 (N=64 -> waves 0..3) ----
    if (wave < 4) {
        const int ncol = wave * 16 + m;
        const __bf16* wrow = W2t + (size_t)ncol * 256;
        v8f acc = {};
        #pragma unroll
        for (int kk = 0; kk < 8; ++kk) {
            const int k0 = kk * 32;
            v8bf alo = *(const v8bf*)&Hs[m][k0 + hi * 8];
            v8bf ahi = *(const v8bf*)&Hs[m][k0 + 16 + hi * 8];
            v16bf a = __builtin_shufflevector(alo, ahi,
                0,1,2,3,4,5,6,7,8,9,10,11,12,13,14,15);
            v16bf b = __builtin_shufflevector(*(const v8bf*)(wrow + k0 + hi * 8),
                *(const v8bf*)(wrow + k0 + 16 + hi * 8),
                0,1,2,3,4,5,6,7,8,9,10,11,12,13,14,15);
            acc = __builtin_amdgcn_wmma_f32_16x16x32_bf16(
                false, a, false, b, (short)0, acc, false, false);
        }
        const float bias = b2[ncol];
        #pragma unroll
        for (int j = 0; j < 8; ++j) {
            float v = acc[j] + bias;
            z[(size_t)(row0 + j + 8 * hi) * 64 + ncol] = (v > 0.f ? v : 0.f);
        }
    }
}

// ---------------- out = beta*out; item region additionally += z -------------
__global__ __launch_bounds__(256) void final_kernel(
    float* __restrict__ out, const float* __restrict__ z,
    float beta, size_t nu_elems, size_t total)
{
    size_t t = (size_t)blockIdx.x * blockDim.x + threadIdx.x;
    if (t >= total) return;
    float v = out[t] * beta;
    if (t >= nu_elems) v += z[t - nu_elems];
    out[t] = v;
}

static inline int blocks_for(size_t n, int bs) { return (int)((n + bs - 1) / bs); }

extern "C" void kernel_launch(void* const* d_in, const int* in_sizes, int n_in,
                              void* d_out, int out_size, void* d_ws, size_t ws_size,
                              hipStream_t stream) {
    const float* user_emb = (const float*)d_in[0];
    const float* item_emb = (const float*)d_in[1];
    const float* feat0    = (const float*)d_in[2];
    const float* feat1    = (const float*)d_in[3];
    const float* g_w      = (const float*)d_in[4];
    const float* ig0_w    = (const float*)d_in[5];
    const float* ig1_w    = (const float*)d_in[6];
    const float* W1       = (const float*)d_in[7];
    const float* b1       = (const float*)d_in[8];
    const float* W2       = (const float*)d_in[9];
    const float* b2       = (const float*)d_in[10];
    const int*   g_src    = (const int*)d_in[11];
    const int*   g_dst    = (const int*)d_in[12];
    const int*   ig0_src  = (const int*)d_in[13];
    const int*   ig0_dst  = (const int*)d_in[14];
    const int*   ig1_src  = (const int*)d_in[15];
    const int*   ig1_dst  = (const int*)d_in[16];
    float* out = (float*)d_out;

    const float BETA = 0.9f, ALPHA = 0.1f;
    const float AB = ALPHA / BETA;      // accumulator init scale
    const int BIG = 0x7fffffff;
    // gamma(k) = beta^k + alpha*(1-beta^k)/(1-beta) == 1 for these params.

    // ---- workspace layout ----
    float* buf0 = (float*)d_ws;                         // NI*128
    float* buf1 = buf0 + (size_t)NI_ * DF_;             // NI*128
    float* buf2 = buf1 + (size_t)NI_ * DF_;             // NI*128
    float* g0   = buf2 + (size_t)NI_ * DF_;             // NN*64 (also hosts W1t/W2t pre-phase-D)
    float* wpg  = g0   + (size_t)NN_ * D_;              // E
    float* wp0  = wpg + E_;                             // EI
    float* wp1  = wp0 + EI_;                            // EI
    float* zbuf = buf0;                                 // NI*64, reused after phase B
    __bf16* W1t = (__bf16*)g0;                          // 256*256 bf16 (128KB)
    __bf16* W2t = W1t + 256 * 256;                      // 256*64 bf16  (32KB)

    const size_t itN   = (size_t)NI_ * (DF_ / 4);       // init threads, item bufs
    const size_t gN    = (size_t)NN_ * (D_ / 4);        // init threads, bipartite
    const size_t esc64 = (size_t)E_ * 16;               // scatter threads, bipartite
    const int    b128  = blocks_for(EI_, 8);            // scatter128: wave per edge

    // ---- preprocessing: w' = beta*w, transposed bf16 weights ----
    scalew_kernel<<<blocks_for(E_, 256), 256, 0, stream>>>(g_w,   wpg, BETA, E_);
    scalew_kernel<<<blocks_for(EI_, 256), 256, 0, stream>>>(ig0_w, wp0, BETA, EI_);
    scalew_kernel<<<blocks_for(EI_, 256), 256, 0, stream>>>(ig1_w, wp1, BETA, EI_);
    wtrans_kernel<<<blocks_for(256 * 256, 256), 256, 0, stream>>>(W1, W1t, 256, 256);
    wtrans_kernel<<<blocks_for(256 * 64, 256), 256, 0, stream>>>(W2, W2t, 256, 64);

    auto hop128 = [&](const float* hsrc, const float* h0, float* agg,
                      const int* es, const int* ed, const float* ew) {
        init_kernel<DF_><<<blocks_for(itN, 256), 256, 0, stream>>>(
            agg, h0, h0, BIG, NI_, AB);
        scatter128_kernel<<<b128, 256, 0, stream>>>(hsrc, es, ed, ew, agg, EI_);
    };

    // ---- phase A: modality 0, 4 hops (unscaled state; result B4 -> buf1) ----
    hop128(feat0, feat0, buf0, ig0_src, ig0_dst, ig0_w);  // hop1: original w
    hop128(buf0,  feat0, buf1, ig0_src, ig0_dst, wp0);
    hop128(buf1,  feat0, buf0, ig0_src, ig0_dst, wp0);
    hop128(buf0,  feat0, buf1, ig0_src, ig0_dst, wp0);

    // ---- phase B: modality 1, 2 hops (result B2 -> buf2) ----
    hop128(feat1, feat1, buf0, ig1_src, ig1_dst, ig1_w);
    hop128(buf0,  feat1, buf2, ig1_src, ig1_dst, wp1);

    // ---- phase C: bf16-WMMA MLP; h = beta*B folded into staging ----
    mlp_kernel<<<NI_ / 16, 512, 0, stream>>>(buf1, buf2, BETA,
                                             W1t, b1, W2t, b2, zbuf);

    // ---- phase D: bipartite graph, 2 hops; hop2 lands in d_out ----
    init_kernel<D_><<<blocks_for(gN, 256), 256, 0, stream>>>(
        g0, user_emb, item_emb, NU_, NN_, AB);
    scatter64_kernel<<<blocks_for(esc64, 256), 256, 0, stream>>>(
        user_emb, item_emb, NU_, g_src, g_dst, g_w, g0, E_);   // hop1: original w

    init_kernel<D_><<<blocks_for(gN, 256), 256, 0, stream>>>(
        out, user_emb, item_emb, NU_, NN_, AB);
    scatter64_kernel<<<blocks_for(esc64, 256), 256, 0, stream>>>(
        g0, g0, BIG, g_src, g_dst, wpg, out, E_);

    // ---- final: out = beta*out, items += z ----
    const size_t total = (size_t)NN_ * D_;
    final_kernel<<<blocks_for(total, 256), 256, 0, stream>>>(
        out, zbuf, BETA, (size_t)NU_ * D_, total);
}
